// CNN_BiMAMBA_MHA_AV_42288247997048
// MI455X (gfx1250) — matrix operand order repaired and checked by reading, not compiled, not run on hardware
//
#include <hip/hip_runtime.h>
#include <math.h>

// ---------------------------------------------------------------------------
// CDNA5 (gfx1250) CNN + BiMamba + MHA AV model, WMMA bf16 throughout.
//  - All GEMM operands are bf16 in memory; producers convert once.
//  - Activation buffers are K-padded to 32 (zero-filled); packed weights are
//    K-padded to 32 AND N-padded to 64 (zero rows) -> the hot GEMM loop is a
//    single basic block: 2x b128 (A) + 4x (2x b128 (B) + v_wmma), no guards.
//  - Each wave computes 16x64 (4 accumulators); 8 waves / 256-thread block.
//  - Packed weights (~6MB) are L2-resident (192MB L2): no LDS staging needed.
//  - Scan / dwconv / rmsnorm / softmax stay VALU (sequential / tiny).
// ---------------------------------------------------------------------------

#define DEV __device__ __forceinline__

typedef __attribute__((ext_vector_type(16))) __bf16  v16bf;
typedef __attribute__((ext_vector_type(8)))  float   v8f;
typedef __attribute__((ext_vector_type(4)))  unsigned u32x4;

union Frag { unsigned short u[16]; u32x4 q[2]; v16bf v; };

DEV unsigned short f2bf(float f) {
  unsigned u = __builtin_bit_cast(unsigned, f);
  unsigned r = u + 0x7FFFu + ((u >> 16) & 1u);   // RNE
  return (unsigned short)(r >> 16);
}
DEV float bf2f(unsigned short h) {
  unsigned u = ((unsigned)h) << 16;
  return __builtin_bit_cast(float, u);
}
DEV float dev_silu(float x) { return x / (1.f + __expf(-x)); }

// ---------------------------------------------------------------------------
// GEMM: C = act(alpha * A(MxK) * B + bias), M % 16 == 0 at every call site.
// AMODE 0: A f32 rows, per-element + K guard      (dt projection, K=16/4)
// AMODE 1: A bf16 rows, aligned vector loads, width >= round32(K)
// AMODE 2: A bf16 rows, per-element + K guard     (audio scores, hd=15)
// BMODE 0: B packed bf16 (N64-pad x K32-pad) rows -> unconditional loads
// BMODE 1: B bf16 trans rows B[k,n]=Bp[n*ldb+k]; requires N % 64 == 0
// BMODE 2: B bf16 plain B[k,n]=Bp[k*ldb+n]; K%32==0; ghost cols mem-safe
// Outputs: Cf (f32) and/or Cb (bf16); store guards only in epilogue.
// ---------------------------------------------------------------------------
struct GArg {
  const void* A; const void* B; const float* bias;
  float* Cf; unsigned short* Cb;
  int M, N, K, lda, ldb, ldc, act, inner;
  float alpha;
  long long sAo, sAi, sBo, sBi, sCo, sCi;
};

template <int AMODE, int BMODE>
__global__ void __launch_bounds__(256) k_gemm(GArg g) {
  const int wave = threadIdx.x >> 5, lane = threadIdx.x & 31;
  const int m0 = (blockIdx.x * 8 + wave) * 16;
  if (m0 >= g.M) return;                        // wave-uniform
  const int n0 = blockIdx.y * 64;
  const int z  = blockIdx.z;
  const int mr = lane & 15, kh = lane >> 4;
  const long long Ab = (long long)(z / g.inner) * g.sAo + (long long)(z % g.inner) * g.sAi;
  const long long Bb = (long long)(z / g.inner) * g.sBo + (long long)(z % g.inner) * g.sBi;
  const long long Cbo = (long long)(z / g.inner) * g.sCo + (long long)(z % g.inner) * g.sCi;

  v8f acc[4];
#pragma unroll
  for (int j = 0; j < 4; j++) acc[j] = (v8f){0.f,0.f,0.f,0.f,0.f,0.f,0.f,0.f};

  const int Kp = (g.K + 31) & ~31;
  for (int k0 = 0; k0 < Kp; k0 += 32) {
    Frag a;
    if (AMODE == 1) {
      const unsigned short* ap =
          (const unsigned short*)g.A + Ab + (long long)(m0 + mr) * g.lda + k0 + kh * 16;
      a.q[0] = *(const u32x4*)ap;
      a.q[1] = *(const u32x4*)(ap + 8);
    } else if (AMODE == 0) {
      const float* ar = (const float*)g.A + Ab + (long long)(m0 + mr) * g.lda;
#pragma unroll
      for (int i = 0; i < 16; i++) {
        int kk = k0 + kh * 16 + i;
        a.u[i] = (kk < g.K) ? f2bf(ar[kk]) : (unsigned short)0;
      }
    } else {  // AMODE 2
      const unsigned short* ap =
          (const unsigned short*)g.A + Ab + (long long)(m0 + mr) * g.lda;
#pragma unroll
      for (int i = 0; i < 16; i++) {
        int kk = k0 + kh * 16 + i;
        a.u[i] = (kk < g.K) ? ap[kk] : (unsigned short)0;
      }
    }
#pragma unroll
    for (int j = 0; j < 4; j++) {
      const int col = n0 + j * 16 + mr;
      Frag b;
      if (BMODE == 0) {        // N-padded packed weights: always in-bounds
        const unsigned short* bp =
            (const unsigned short*)g.B + (long long)col * g.ldb + k0 + kh * 16;
        b.q[0] = ((const u32x4*)bp)[0];
        b.q[1] = ((const u32x4*)bp)[1];
      } else if (BMODE == 1) { // N % 64 == 0 guaranteed by caller
        const unsigned short* bp =
            (const unsigned short*)g.B + Bb + (long long)col * g.ldb;
        if (k0 + 32 <= g.K) {
          b.q[0] = ((const u32x4*)(bp + k0 + kh * 16))[0];
          b.q[1] = ((const u32x4*)(bp + k0 + kh * 16))[1];
        } else {
#pragma unroll
          for (int i = 0; i < 16; i++) {
            int kk = k0 + kh * 16 + i;
            b.u[i] = (kk < g.K) ? bp[kk] : (unsigned short)0;
          }
        }
      } else {                 // BMODE 2: ghost cols memory-safe in qkv row
        const unsigned short* bp = (const unsigned short*)g.B + Bb + col;
#pragma unroll
        for (int i = 0; i < 16; i++)
          b.u[i] = bp[(long long)(k0 + kh * 16 + i) * g.ldb];
      }
      acc[j] = __builtin_amdgcn_wmma_f32_16x16x32_bf16(false, a.v, false, b.v,
                                                       (short)0, acc[j], false, false);
    }
  }
#pragma unroll
  for (int j = 0; j < 4; j++) {
    const int col = n0 + j * 16 + mr;
    if (col >= g.N) continue;
    float bv = g.bias ? g.bias[col] : 0.f;
#pragma unroll
    for (int r = 0; r < 8; r++) {
      int rr = m0 + kh * 8 + r;
      float v = acc[j][r] * g.alpha + bv;
      if (g.act == 1)      v = fmaxf(v, 0.f);
      else if (g.act == 2) v = (v > 20.f) ? v : log1pf(__expf(v));
      long long ci = Cbo + (long long)rr * g.ldc + col;
      if (g.Cf) g.Cf[ci] = v;
      if (g.Cb) g.Cb[ci] = f2bf(v);
    }
  }
}

// ---------------------------------------------------------------------------
// Implicit-GEMM 3x3 stride-2 SAME conv (pad lo=0, hi=1), bias+relu fused.
// A: im2col gather (INBF: 0 = f32 video input, 1 = bf16 activations)
// B: packed bf16 (OC64-pad, Kp) -> unconditional loads.  Output NCHW bf16.
// ---------------------------------------------------------------------------
struct CArg {
  const void* in;
  const unsigned short* w;
  const float* bias;
  unsigned short* out;
  int NF, CIN, IH, IW, OC, OH, OW, Kp;
};

template <int INBF>
__global__ void __launch_bounds__(256) k_conv(CArg c) {
  const int OHW = c.OH * c.OW;
  const int M = c.NF * OHW, K = c.CIN * 9;
  const int wave = threadIdx.x >> 5, lane = threadIdx.x & 31;
  const int m0 = (blockIdx.x * 8 + wave) * 16;
  if (m0 >= M) return;
  const int n0 = blockIdx.y * 64;
  const int mr = lane & 15, kh = lane >> 4;
  const int row = m0 + mr;
  const int frame = row / OHW, pos = row % OHW;
  const int oy = pos / c.OW, ox = pos % c.OW;

  v8f acc[4];
#pragma unroll
  for (int j = 0; j < 4; j++) acc[j] = (v8f){0.f,0.f,0.f,0.f,0.f,0.f,0.f,0.f};

  for (int k0 = 0; k0 < c.Kp; k0 += 32) {
    Frag a;
#pragma unroll
    for (int i = 0; i < 16; i++) {
      int kk = k0 + kh * 16 + i;
      unsigned short vv = 0;
      if (kk < K) {
        int ic = kk / 9, rr2 = kk % 9, ky = rr2 / 3, kx = rr2 % 3;
        int iy = oy * 2 + ky, ix = ox * 2 + kx;
        if (iy < c.IH && ix < c.IW) {
          long long idx = ((long long)(frame * c.CIN + ic) * c.IH + iy) * c.IW + ix;
          vv = INBF ? ((const unsigned short*)c.in)[idx]
                    : f2bf(((const float*)c.in)[idx]);
        }
      }
      a.u[i] = vv;
    }
#pragma unroll
    for (int j = 0; j < 4; j++) {
      const int col = n0 + j * 16 + mr;
      const unsigned short* bp = c.w + (long long)col * c.Kp + k0 + kh * 16;
      Frag b;
      b.q[0] = ((const u32x4*)bp)[0];
      b.q[1] = ((const u32x4*)bp)[1];
      acc[j] = __builtin_amdgcn_wmma_f32_16x16x32_bf16(false, a.v, false, b.v,
                                                       (short)0, acc[j], false, false);
    }
  }
#pragma unroll
  for (int j = 0; j < 4; j++) {
    const int col = n0 + j * 16 + mr;
    if (col >= c.OC) continue;
    float bv = c.bias[col];
#pragma unroll
    for (int r = 0; r < 8; r++) {
      int rr = m0 + kh * 8 + r;
      int fr = rr / OHW, p2 = rr % OHW, y = p2 / c.OW, x = p2 % c.OW;
      float v = fmaxf(acc[j][r] + bv, 0.f);
      c.out[((long long)(fr * c.OC + col) * c.OH + y) * c.OW + x] = f2bf(v);
    }
  }
}

// --------------------------- weight packing --------------------------------
// bf16 (Np x Kp), zero pad in both K (to 32) and N (to 64). trans=1: W=(K,N).
__global__ void k_wprep(const float* W, unsigned short* o, int K, int N,
                        int Kp, int Np, int trans) {
  int id = blockIdx.x * blockDim.x + threadIdx.x;
  if (id >= Np * Kp) return;
  int n = id / Kp, k = id % Kp;
  float v = 0.f;
  if (k < K && n < N) v = trans ? W[(long long)k * N + n] : W[(long long)n * K + k];
  o[id] = f2bf(v);
}

// ----------------------------- small kernels -------------------------------
__global__ void k_pool(const unsigned short* in, unsigned short* out, int n, int HW) {
  int id = blockIdx.x * blockDim.x + threadIdx.x;
  if (id >= n) return;
  const unsigned short* p = in + (long long)id * HW;
  float s = 0.f;
  for (int i = 0; i < HW; i++) s += bf2f(p[i]);
  out[id] = f2bf(s / (float)HW);
}

__global__ void k_rms(const float* x, const float* w, unsigned short* y,
                      int rows, int d, int Kp) {
  int r = blockIdx.x * blockDim.x + threadIdx.x;
  if (r >= rows) return;
  const float* xr = x + (long long)r * d;
  float s = 0.f;
  for (int i = 0; i < d; i++) s += xr[i] * xr[i];
  float inv = rsqrtf(s / (float)d + 1e-5f);
  unsigned short* yr = y + (long long)r * Kp;
  for (int i = 0; i < Kp; i++)
    yr[i] = (i < d) ? f2bf(xr[i] * inv * w[i]) : (unsigned short)0;
}

__global__ void k_dwconv(const float* uz, int lduz, const float* w, const float* bias,
                         float* uc, unsigned short* ucb, int Bn, int L, int ED, int KpED) {
  int id = blockIdx.x * blockDim.x + threadIdx.x;
  if (id >= Bn * L * KpED) return;
  int row = id / KpED, ch = id % KpED;
  if (ch >= ED) { ucb[(long long)row * KpED + ch] = 0; return; }
  int b = row / L, t = row % L;
  float s = bias[ch];
#pragma unroll
  for (int j = 0; j < 4; j++) {
    int ti = t - 3 + j;
    if (ti >= 0) s += w[ch * 4 + j] * uz[(long long)(b * L + ti) * lduz + ch];
  }
  float v = dev_silu(s);
  uc[(long long)row * ED + ch] = v;
  ucb[(long long)row * KpED + ch] = f2bf(v);
}

__global__ void k_scan(const float* uc, const float* dtb, const float* dbc,
                       const float* A_log, const float* Dp,
                       const float* uz, int zoff, int lduz,
                       unsigned short* yszb, int Bn, int L, int ED, int KpED, int R) {
  int id = blockIdx.x * blockDim.x + threadIdx.x;
  if (id >= Bn * KpED) return;
  int b = id / KpED, ch = id % KpED;
  if (ch >= ED) {
    for (int t = 0; t < L; t++) yszb[((long long)b * L + t) * KpED + ch] = 0;
    return;
  }
  float a[16], h[16];
#pragma unroll
  for (int n = 0; n < 16; n++) { a[n] = -__expf(A_log[ch * 16 + n]); h[n] = 0.f; }
  float Dch = Dp[ch];
  int ldx = R + 32;
  for (int t = 0; t < L; t++) {
    long long row = (long long)b * L + t;
    float u  = uc[row * ED + ch];
    float dt = dtb[row * ED + ch];
    const float* bc = dbc + row * ldx;
    float y = 0.f;
#pragma unroll
    for (int n = 0; n < 16; n++) {
      float dA = __expf(dt * a[n]);
      h[n] = dA * h[n] + dt * bc[R + n] * u;
      y += h[n] * bc[R + 16 + n];
    }
    float zv = uz[row * lduz + zoff + ch];
    yszb[row * KpED + ch] = f2bf((y + u * Dch) * dev_silu(zv));
  }
}

__global__ void k_add(float* x, const float* d, int n) {
  int i = blockIdx.x * blockDim.x + threadIdx.x;
  if (i < n) x[i] += d[i];
}
__global__ void k_copy(const float* in, float* out, int n) {
  int i = blockIdx.x * blockDim.x + threadIdx.x;
  if (i < n) out[i] = in[i];
}
__global__ void k_zerob(unsigned short* p, int n) {
  int i = blockIdx.x * blockDim.x + threadIdx.x;
  if (i < n) p[i] = 0;
}
__global__ void k_flip(const float* in, float* out, int Bn, int L, int D) {
  int i = blockIdx.x * blockDim.x + threadIdx.x;
  if (i >= Bn * L * D) return;
  int b = i / (L * D), rem = i % (L * D), t = rem / D, c = rem % D;
  out[((long long)b * L + t) * D + c] = in[((long long)b * L + (L - 1 - t)) * D + c];
}
__global__ void k_concat(const float* a, const float* b, unsigned short* out,
                         int rows, int d1, int d2, int Kp) {
  int i = blockIdx.x * blockDim.x + threadIdx.x;
  if (i >= rows * Kp) return;
  int r = i / Kp, c = i % Kp;
  float v = 0.f;
  if (c < d1)            v = a[(long long)r * d1 + c];
  else if (c < d1 + d2)  v = b[(long long)r * d2 + (c - d1)];
  out[i] = f2bf(v);
}
__global__ void k_softmax(const float* a, unsigned short* ob, int rows, int L) {
  int r = blockIdx.x * blockDim.x + threadIdx.x;
  if (r >= rows) return;
  const float* p = a + (long long)r * L;
  unsigned short* q = ob + (long long)r * L;
  float m = -1e30f;
  for (int i = 0; i < L; i++) m = fmaxf(m, p[i]);
  float s = 0.f;
  for (int i = 0; i < L; i++) s += __expf(p[i] - m);
  float inv = 1.f / s;
  for (int i = 0; i < L; i++) q[i] = f2bf(__expf(p[i] - m) * inv);
}
__global__ void k_head(const float* vmha, const float* amha,
                       const float* w, const float* bias, float* out) {
  int t = threadIdx.x;
  if (t >= 16) return;
  int b = t >> 1, o = t & 1;
  const float* vr = vmha + ((long long)b * 64 + 63) * 512;
  const float* ar = amha + ((long long)b * 256 + 255) * 120;
  const float* wr = w + o * 632;
  float s = bias[o];
  for (int i = 0; i < 512; i++) s += vr[i] * wr[i];
  for (int i = 0; i < 120; i++) s += ar[i] * wr[512 + i];
  out[b * 2 + o] = s;
}

// ----------------------------- host helpers --------------------------------
static inline int round32(int k) { return (k + 31) & ~31; }
static inline int round64(int k) { return (k + 63) & ~63; }

static void gemm(hipStream_t s, int amode, int bmode,
                 const void* A, const void* B, const float* bias,
                 float* Cf, unsigned short* Cb,
                 int M, int N, int K, int lda, int ldb, int ldc,
                 float alpha = 1.f, int act = 0,
                 int nbatch = 1, int inner = 1,
                 long long sAo = 0, long long sAi = 0,
                 long long sBo = 0, long long sBi = 0,
                 long long sCo = 0, long long sCi = 0) {
  GArg g;
  g.A = A; g.B = B; g.bias = bias; g.Cf = Cf; g.Cb = Cb;
  g.M = M; g.N = N; g.K = K; g.lda = lda; g.ldb = ldb; g.ldc = ldc;
  g.act = act; g.inner = inner; g.alpha = alpha;
  g.sAo = sAo; g.sAi = sAi; g.sBo = sBo; g.sBi = sBi; g.sCo = sCo; g.sCi = sCi;
  dim3 grd((unsigned)((M / 16 + 7) / 8), (unsigned)((N + 63) / 64), (unsigned)nbatch);
  dim3 blk(256);
  if      (amode == 1 && bmode == 0) k_gemm<1, 0><<<grd, blk, 0, s>>>(g);
  else if (amode == 0 && bmode == 0) k_gemm<0, 0><<<grd, blk, 0, s>>>(g);
  else if (amode == 1 && bmode == 1) k_gemm<1, 1><<<grd, blk, 0, s>>>(g);
  else if (amode == 2 && bmode == 1) k_gemm<2, 1><<<grd, blk, 0, s>>>(g);
  else                               k_gemm<1, 2><<<grd, blk, 0, s>>>(g);
}

struct MLay {
  const float *norm, *conv_w, *conv_b, *dt_b, *A_log, *D;
  const unsigned short *in_projT, *x_projT, *dt_wT, *out_projT;
};
struct MBufs {
  unsigned short *xnb, *ucb, *yszb;
  float *uz, *uc, *dbc, *dtb, *delta;
};

static void mamba_run(hipStream_t s, float* x, const MLay* lay,
                      int Bn, int L, int d, const MBufs& b) {
  int ED = 2 * d, R = (d + 15) / 16, BL = Bn * L;
  int KpD = round32(d), KpED = round32(ED);
  for (int l = 0; l < 2; l++) {
    const MLay& p = lay[l];
    k_rms<<<(BL + 255) / 256, 256, 0, s>>>(x, p.norm, b.xnb, BL, d, KpD);
    gemm(s, 1, 0, b.xnb, p.in_projT, nullptr, b.uz, nullptr,
         BL, 2 * ED, d, KpD, KpD, 2 * ED);
    int n1 = BL * KpED;
    k_dwconv<<<(n1 + 255) / 256, 256, 0, s>>>(b.uz, 2 * ED, p.conv_w, p.conv_b,
                                              b.uc, b.ucb, Bn, L, ED, KpED);
    gemm(s, 1, 0, b.ucb, p.x_projT, nullptr, b.dbc, nullptr,
         BL, R + 32, ED, KpED, KpED, R + 32);
    gemm(s, 0, 0, b.dbc, p.dt_wT, p.dt_b, b.dtb, nullptr,
         BL, ED, R, R + 32, 32, ED, 1.f, /*softplus*/2);
    int nsc = Bn * KpED;
    k_scan<<<(nsc + 127) / 128, 128, 0, s>>>(b.uc, b.dtb, b.dbc, p.A_log, p.D,
                                             b.uz, ED, 2 * ED, b.yszb, Bn, L, ED, KpED, R);
    gemm(s, 1, 0, b.yszb, p.out_projT, nullptr, b.delta, nullptr,
         BL, d, ED, KpED, KpED, d);
    int na = BL * d;
    k_add<<<(na + 255) / 256, 256, 0, s>>>(x, b.delta, na);
  }
}

static void mha_run(hipStream_t s, const unsigned short* xb,
                    const unsigned short* in_wT, const float* in_b,
                    const unsigned short* out_wT, const float* out_b,
                    int Bn, int L, int E, int nh,
                    unsigned short* qkvb, float* attn, unsigned short* attnb,
                    unsigned short* obufb, float* out) {
  int BL = Bn * L, hd = E / nh, KpE = round32(E);
  gemm(s, 1, 0, xb, in_wT, in_b, nullptr, qkvb, BL, 3 * E, E, KpE, KpE, 3 * E);
  float sc = 1.f / sqrtf((float)hd);
  // scores[b,h] = q @ k^T / sqrt(hd)  (N = L, multiple of 64)
  gemm(s, (hd % 8 == 0) ? 1 : 2, 1, qkvb, qkvb + E, nullptr, attn, nullptr,
       L, L, hd, 3 * E, 3 * E, L, sc, 0,
       Bn * nh, nh,
       (long long)L * 3 * E, hd, (long long)L * 3 * E, hd,
       (long long)nh * L * L, (long long)L * L);
  int rows = Bn * nh * L;
  k_softmax<<<(rows + 255) / 256, 256, 0, s>>>(attn, attnb, rows, L);
  int nzb = BL * KpE;
  k_zerob<<<(nzb + 255) / 256, 256, 0, s>>>(obufb, nzb);
  gemm(s, 1, 2, attnb, qkvb + 2 * E, nullptr, nullptr, obufb,
       L, hd, L, L, 3 * E, KpE, 1.f, 0,
       Bn * nh, nh,
       (long long)nh * L * L, (long long)L * L, (long long)L * 3 * E, hd,
       (long long)L * KpE, hd);
  gemm(s, 1, 0, obufb, out_wT, out_b, out, nullptr, BL, E, E, KpE, KpE, E);
}

// ---------------------------------------------------------------------------
// Input index map (setup_inputs dict insertion order, depth-first):
// 0 video  1 audio
// 2..9 cnn: c1_w c1_b c2_w c2_b c3_w c3_b fc_w fc_b
// 10..29 v_fwd (2 layers x 10), 30..49 v_bwd, 50..69 a_fwd, 70..89 a_bwd
// 90..93 v_mha, 94..97 a_mha, 98 fc_w, 99 fc_b
// ---------------------------------------------------------------------------
extern "C" void kernel_launch(void* const* d_in, const int* in_sizes, int n_in,
                              void* d_out, int out_size, void* d_ws, size_t ws_size,
                              hipStream_t stream) {
  (void)in_sizes; (void)n_in; (void)out_size; (void)ws_size;
  const float* video = (const float*)d_in[0];
  const float* audio = (const float*)d_in[1];
  auto P = [&](int i) -> const float* { return (const float*)d_in[i]; };

  size_t off = 0;
  auto alloc = [&](size_t bytes) -> void* {
    void* p = (char*)d_ws + off;
    off = (off + bytes + 255) & ~(size_t)255;
    return p;
  };
  auto prep = [&](const float* W, int K, int N, int trans) -> const unsigned short* {
    int Kp = round32(K), Np = round64(N);
    unsigned short* o = (unsigned short*)alloc((size_t)Np * Kp * 2);
    int n = Np * Kp;
    k_wprep<<<(n + 255) / 256, 256, 0, stream>>>(W, o, K, N, Kp, Np, trans);
    return o;
  };

  // activations / scratch
  unsigned short* conv1b = (unsigned short*)alloc((size_t)64 * 32 * 56 * 56 * 2);
  unsigned short* conv2b = (unsigned short*)alloc((size_t)64 * 64 * 28 * 28 * 2);
  unsigned short* conv3b = (unsigned short*)alloc((size_t)64 * 128 * 14 * 14 * 2);
  unsigned short* pooledb = (unsigned short*)alloc((size_t)512 * 128 * 2);
  float* xv    = (float*)alloc((size_t)512 * 256 * 4);
  float* xwork = (float*)alloc((size_t)131072 * 4);
  float* vf = (float*)alloc((size_t)512 * 256 * 4);
  float* vb = (float*)alloc((size_t)512 * 256 * 4);
  float* af = (float*)alloc((size_t)2048 * 60 * 4);
  float* ab = (float*)alloc((size_t)2048 * 60 * 4);
  MBufs mb;
  mb.xnb   = (unsigned short*)alloc((size_t)131072 * 2);
  mb.ucb   = (unsigned short*)alloc((size_t)262144 * 2);
  mb.yszb  = (unsigned short*)alloc((size_t)262144 * 2);
  mb.uz    = (float*)alloc((size_t)524288 * 4);
  mb.uc    = (float*)alloc((size_t)262144 * 4);
  mb.dbc   = (float*)alloc((size_t)73728 * 4);
  mb.dtb   = (float*)alloc((size_t)262144 * 4);
  mb.delta = (float*)alloc((size_t)131072 * 4);
  unsigned short* catb  = (unsigned short*)alloc((size_t)262144 * 2);
  unsigned short* qkvb  = (unsigned short*)alloc((size_t)786432 * 2);
  float*          attn  = (float*)alloc((size_t)4194304 * 4);
  unsigned short* attnb = (unsigned short*)alloc((size_t)4194304 * 2);
  unsigned short* obufb = (unsigned short*)alloc((size_t)262144 * 2);
  float* vmha = (float*)alloc((size_t)512 * 512 * 4);
  float* amha = (float*)alloc((size_t)2048 * 120 * 4);

  // ---------------- pack all weights (bf16, K->32 pad, N->64 pad) ----------
  const unsigned short* wc1 = prep(P(2), 27,  32,  0);
  const unsigned short* wc2 = prep(P(4), 288, 64,  0);
  const unsigned short* wc3 = prep(P(6), 576, 128, 0);
  const unsigned short* wfc = prep(P(8), 128, 256, 0);

  MLay lay[4][2];
  const int bases[4] = {10, 30, 50, 70};
  const int dms[4]   = {256, 256, 60, 60};
  for (int m = 0; m < 4; m++) {
    int d = dms[m], ED = 2 * d, R = (d + 15) / 16;
    for (int l = 0; l < 2; l++) {
      int b0 = bases[m] + 10 * l;
      MLay& L2 = lay[m][l];
      L2.norm   = P(b0 + 0);
      L2.conv_w = P(b0 + 2);
      L2.conv_b = P(b0 + 3);
      L2.dt_b   = P(b0 + 6);
      L2.A_log  = P(b0 + 7);
      L2.D      = P(b0 + 8);
      L2.in_projT  = prep(P(b0 + 1), d,  2 * ED, 1);
      L2.x_projT   = prep(P(b0 + 4), ED, R + 32, 1);
      L2.dt_wT     = prep(P(b0 + 5), R,  ED,     1);
      L2.out_projT = prep(P(b0 + 9), ED, d,      1);
    }
  }
  const unsigned short* vin_wT  = prep(P(90), 512, 1536, 0);
  const unsigned short* vout_wT = prep(P(92), 512, 512,  0);
  const unsigned short* ain_wT  = prep(P(94), 120, 360,  0);
  const unsigned short* aout_wT = prep(P(96), 120, 120,  0);

  // ---------------- video CNN (chunked implicit-GEMM WMMA) -----------------
  for (int c = 0; c < 8; c++) {
    const float* vin = video + (long long)c * 64 * 3 * 112 * 112;
    {
      CArg a{vin, wc1, P(3), conv1b, 64, 3, 112, 112, 32, 56, 56, 32};
      int M = 64 * 56 * 56;
      dim3 g((unsigned)((M / 16 + 7) / 8), 1, 1);
      k_conv<0><<<g, dim3(256), 0, stream>>>(a);
    }
    {
      CArg a{conv1b, wc2, P(5), conv2b, 64, 32, 56, 56, 64, 28, 28, 288};
      int M = 64 * 28 * 28;
      dim3 g((unsigned)((M / 16 + 7) / 8), 1, 1);
      k_conv<1><<<g, dim3(256), 0, stream>>>(a);
    }
    {
      CArg a{conv2b, wc3, P(7), conv3b, 64, 64, 28, 28, 128, 14, 14, 576};
      int M = 64 * 14 * 14;
      dim3 g((unsigned)((M / 16 + 7) / 8), 2, 1);
      k_conv<1><<<g, dim3(256), 0, stream>>>(a);
    }
    k_pool<<<(64 * 128 + 255) / 256, 256, 0, stream>>>(conv3b, pooledb + c * 64 * 128,
                                                       64 * 128, 196);
  }
  gemm(stream, 1, 0, pooledb, wfc, P(9), xv, nullptr, 512, 256, 128, 128, 128, 256);

  // ---------------- video bidirectional mamba + MHA ------------------------
  int nv = 512 * 256;
  k_copy<<<(nv + 255) / 256, 256, 0, stream>>>(xv, xwork, nv);
  mamba_run(stream, xwork, lay[0], 8, 64, 256, mb);
  k_copy<<<(nv + 255) / 256, 256, 0, stream>>>(xwork, vf, nv);
  k_flip<<<(nv + 255) / 256, 256, 0, stream>>>(xv, xwork, 8, 64, 256);
  mamba_run(stream, xwork, lay[1], 8, 64, 256, mb);
  k_flip<<<(nv + 255) / 256, 256, 0, stream>>>(xwork, vb, 8, 64, 256);
  k_concat<<<(512 * 512 + 255) / 256, 256, 0, stream>>>(vf, vb, catb, 512, 256, 256, 512);
  mha_run(stream, catb, vin_wT, P(91), vout_wT, P(93), 8, 64, 512, 8,
          qkvb, attn, attnb, obufb, vmha);

  // ---------------- audio bidirectional mamba + MHA ------------------------
  int na = 2048 * 60;
  k_copy<<<(na + 255) / 256, 256, 0, stream>>>(audio, xwork, na);
  mamba_run(stream, xwork, lay[2], 8, 256, 60, mb);
  k_copy<<<(na + 255) / 256, 256, 0, stream>>>(xwork, af, na);
  k_flip<<<(na + 255) / 256, 256, 0, stream>>>(audio, xwork, 8, 256, 60);
  mamba_run(stream, xwork, lay[3], 8, 256, 60, mb);
  k_flip<<<(na + 255) / 256, 256, 0, stream>>>(xwork, ab, 8, 256, 60);
  k_concat<<<(2048 * 128 + 255) / 256, 256, 0, stream>>>(af, ab, catb, 2048, 60, 60, 128);
  mha_run(stream, catb, ain_wT, P(95), aout_wT, P(97), 8, 256, 120, 8,
          qkvb, attn, attnb, obufb, amha);

  // ---------------- final head --------------------------------------------
  k_head<<<1, 32, 0, stream>>>(vmha, amha, P(98), P(99), (float*)d_out);
}
